// MultiHeadSelfAttention_59261958750606
// MI455X (gfx1250) — matrix operand order repaired and checked by reading, not compiled, not run on hardware
//
#include <hip/hip_runtime.h>

// ---------------- problem constants ----------------
#define EMBED   1024
#define HEADS   16
#define HD      64
#define BATCH   2
#define SEQ     2048
#define MROWS   (BATCH * SEQ)   // 4096 rows of x flattened
#define E3      (3 * EMBED)     // 3072 qkv output dim

#define BROW_PAD 1032           // B panel row stride in ushorts (2064B = 129*16B)
#define KROW_PAD 72             // K tile row stride (144B = 9*16B)
#define VROW_PAD 40             // V tile row stride (80B = 5*16B)

// ---------------- WMMA fragment types ----------------
typedef __attribute__((ext_vector_type(16))) __bf16 v16bf;
typedef __attribute__((ext_vector_type(8)))  float  v8f;
typedef __attribute__((ext_vector_type(4)))  unsigned int v4u;
typedef __attribute__((ext_vector_type(4)))  float  v4f;
typedef __attribute__((ext_vector_type(4)))  unsigned short v4us;

union FragB16 {           // 16 bf16 = 32 bytes = 8 VGPRs (A or B fragment)
    v16bf v;
    v4u   q[2];
};

__device__ __forceinline__ unsigned short f2bf(float f) {
    union { float f; unsigned u; } x; x.f = f;
    unsigned u = x.u;
    u += 0x7FFFu + ((u >> 16) & 1u);   // round-to-nearest-even
    return (unsigned short)(u >> 16);
}

// CDNA5 async global->LDS copy, 16B per lane (tracked by ASYNCcnt)
__device__ __forceinline__ void async_b128(const unsigned short* g,
                                           unsigned short* l) {
    unsigned loff = (unsigned)(unsigned long long)l;   // low 32 bits = LDS offset
    asm volatile("global_load_async_to_lds_b128 %0, %1, off"
                 :: "v"(loff), "v"(g)
                 : "memory");
}
template<int N>
__device__ __forceinline__ void wait_async() {
    asm volatile("s_wait_asynccnt %0" :: "n"(N) : "memory");
}
// Compiler-only fence: same-wave LDS ops are executed in order by hardware
// (DScnt, in-order per wave), so wave-private LDS store->load transposes need
// no barrier -- only a scheduling fence so the backend keeps program order.
__device__ __forceinline__ void lds_fence() {
    asm volatile("" ::: "memory");
}

// ---------------- f32 -> bf16 bulk convert (4 elems/thread) ----------------
__global__ void k_cvt_bf16(const float* __restrict__ src,
                           unsigned short* __restrict__ dst, int n4) {
    int i = blockIdx.x * blockDim.x + threadIdx.x;
    if (i < n4) {
        v4f f = *(const v4f*)(src + (size_t)i * 4);
        v4us o;
        o.x = f2bf(f.x); o.y = f2bf(f.y); o.z = f2bf(f.z); o.w = f2bf(f.w);
        *(v4us*)(dst + (size_t)i * 4) = o;
    }
}

// ---------------- shared 128x64 GEMM core, full-K B panel in LDS ----------------
// Block = 256 threads = 8 waves, tile 128(M) x 64(N), K = 1024.
// Entire B panel (64 rows x 1024 k bf16, rows padded to 2064B) = 129KB LDS,
// streamed in by 32 async b128 ops per thread (k-ordered), consumed in 4
// mega-stages of 256 k each with progressive s_wait_asynccnt. Only 4 barriers.
__device__ __forceinline__ void gemm_core_128x64(
    const unsigned short* __restrict__ A,   // [*,1024] bf16 row-major
    const unsigned short* __restrict__ W,   // [Nout,1024] bf16 row-major
    unsigned short* Bp,                     // dynamic LDS, [64][BROW_PAD]
    int m0, int n0, int lane, v8f acc[4])
{
    const int half = lane >> 4;
    const int lm   = lane & 15;
    const int tid  = threadIdx.x;
    const int brow   = tid >> 2;     // 0..63 : B row
    const int bchunk = tid & 3;      // 0..3  : base 16B chunk; +4 per j

    // issue the whole panel, in k order: op j covers k chunk (bchunk + 4j)
    const unsigned short* bsrc = W + (size_t)(n0 + brow) * EMBED + bchunk * 8;
    unsigned short* bdst = Bp + (size_t)brow * BROW_PAD + bchunk * 8;
#pragma unroll
    for (int j = 0; j < 32; ++j)
        async_b128(bsrc + j * 32, bdst + j * 32);

    const unsigned short* arow = A + (size_t)(m0 + lm) * EMBED;
    FragB16 aCur, aNxt;
    aCur.q[0] = *(const v4u*)(arow + half * 8);
    aCur.q[1] = *(const v4u*)(arow + 16 + half * 8);
    aNxt = aCur;

    const unsigned short* bfragBase = Bp + (size_t)lm * BROW_PAD + half * 8;

#pragma unroll
    for (int mega = 0; mega < 4; ++mega) {
        // ops are per-thread in-order: <=32-8*(mega+1) outstanding means this
        // mega's 8 ops have landed; barrier extends that to the whole block.
        if      (mega == 0) wait_async<24>();
        else if (mega == 1) wait_async<16>();
        else if (mega == 2) wait_async<8>();
        else                wait_async<0>();
        __syncthreads();
#pragma unroll
        for (int ks = 0; ks < 8; ++ks) {
            const int k0 = mega * 256 + ks * 32;
            const int kn = k0 + 32;
            if (kn < EMBED) {                       // register-prefetch next A
                aNxt.q[0] = *(const v4u*)(arow + kn + half * 8);
                aNxt.q[1] = *(const v4u*)(arow + kn + 16 + half * 8);
            }
#pragma unroll
            for (int t = 0; t < 4; ++t) {
                FragB16 b;
                const unsigned short* bp =
                    bfragBase + (size_t)t * (16 * BROW_PAD) + k0;
                b.q[0] = *(const v4u*)(bp);
                b.q[1] = *(const v4u*)(bp + 16);
                acc[t] = __builtin_amdgcn_wmma_f32_16x16x32_bf16(
                    false, aCur.v, false, b.v, (short)0, acc[t], false, false);
            }
            aCur = aNxt;
        }
    }
}

// ---------------- QKV GEMM: Y = X @ Wqkv^T + b, scatter to Q/K/Vt ----------------
__global__ void __launch_bounds__(256)
k_qkv_gemm(const unsigned short* __restrict__ X,
           const unsigned short* __restrict__ W,
           const float* __restrict__ bias,     // [3072]
           const float* __restrict__ hmask,    // [16]
           unsigned short* __restrict__ Q,     // [B,H,S,HD]
           unsigned short* __restrict__ K,     // [B,H,S,HD]
           unsigned short* __restrict__ Vt)    // [B,H,HD,S]
{
    extern __shared__ __align__(16) unsigned short Bp[];   // [64][BROW_PAD]

    const int wave = threadIdx.x >> 5;
    const int lane = threadIdx.x & 31;
    const int half = lane >> 4;
    const int lm   = lane & 15;

    const int mBlk = blockIdx.x & 31;        // 4096/128 = 32
    const int nBlk = blockIdx.x >> 5;        // 3072/64  = 48
    const int m0 = mBlk * 128 + wave * 16;
    const int n0 = nBlk * 64;

    v8f acc[4] = {};
    gemm_core_128x64(X, W, Bp, m0, n0, lane, acc);

#pragma unroll
    for (int t = 0; t < 4; ++t) {
        const int e     = n0 + t * 16 + lm;   // output feature
        const int h     = e / 192;            // e = h*192 + which*64 + d
        const int rem   = e % 192;
        const int which = rem / 64;
        const int d     = rem % 64;
        const float bv  = bias[e];
        const float hm  = hmask[h];
#pragma unroll
        for (int r = 0; r < 8; ++r) {
            const int m = m0 + r + 8 * half;
            const int n = m >> 11;
            const int s = m & (SEQ - 1);
            const unsigned short o = f2bf((acc[t][r] + bv) * hm);
            const size_t nh = (size_t)n * HEADS + h;
            if (which == 0)      Q [(nh * SEQ + s) * HD + d] = o;
            else if (which == 1) K [(nh * SEQ + s) * HD + d] = o;
            else                 Vt[(nh * HD + d) * SEQ + s] = o;
        }
    }
}

// ---------------- Flash attention ----------------
// grid: (B*H) * (SEQ/64) blocks, 128 threads (4 waves), wave = 16 query rows.
// K/V tiles staged cooperatively in LDS via async loads, double-buffered;
// all 4 waves share them. Only ONE barrier per 32-key step (for the shared
// K/V buffers); the P-tile transpose uses wave-private LDS + in-order DS.
__global__ void __launch_bounds__(128)
k_attention(const unsigned short* __restrict__ Q,
            const unsigned short* __restrict__ K,
            const unsigned short* __restrict__ Vt,
            unsigned short* __restrict__ AO)     // [B,S,EMBED] bf16
{
    __shared__ __align__(16) unsigned short Ks[2][32][KROW_PAD];
    __shared__ __align__(16) unsigned short Vs[2][64][VROW_PAD];
    __shared__ __align__(16) unsigned short pbuf[4][16][32];

    const int wave = threadIdx.x >> 5;
    const int lane = threadIdx.x & 31;
    const int half = lane >> 4;
    const int lm   = lane & 15;
    const int tid  = threadIdx.x;

    const int nh    = blockIdx.x >> 5;       // SEQ/64 = 32 q-tiles per head
    const int qTile = blockIdx.x & 31;
    const int qBase = qTile * 64 + wave * 16;
    const int n = nh >> 4, h = nh & 15;

    const unsigned short* kpane = K + (size_t)nh * SEQ * HD;
    const unsigned short* vpane = Vt + (size_t)nh * HD * SEQ;

    // cooperative stage mapping: 4 async b128 per thread per 32-key stage
    const int krow = tid >> 2;            // 0..31
    const int kc2  = (tid & 3) * 2;       // chunks kc2, kc2+1 of 8
    const int vrow = tid >> 1;            // 0..63
    const int vc2  = (tid & 1) * 2;       // chunks vc2, vc2+1 of 4

    auto issue_stage = [&](int kb, int buf) {
        const unsigned short* ksrc = kpane + (size_t)(kb + krow) * HD + kc2 * 8;
        async_b128(ksrc,     &Ks[buf][krow][kc2 * 8]);
        async_b128(ksrc + 8, &Ks[buf][krow][kc2 * 8 + 8]);
        const unsigned short* vsrc = vpane + (size_t)vrow * SEQ + kb + vc2 * 8;
        async_b128(vsrc,     &Vs[buf][vrow][vc2 * 8]);
        async_b128(vsrc + 8, &Vs[buf][vrow][vc2 * 8 + 8]);
    };

    // Q fragments for hd=64 (two K=32 chunks), loaded once per wave
    FragB16 aQ[2];
    const unsigned short* qrow = Q + ((size_t)nh * SEQ + qBase + lm) * HD;
#pragma unroll
    for (int c = 0; c < 2; ++c) {
        aQ[c].q[0] = *(const v4u*)(qrow + c * 32 + half * 16);
        aQ[c].q[1] = *(const v4u*)(qrow + c * 32 + half * 16 + 8);
    }

    v8f o[4] = {};
    float mrun[8], lrun[8];
#pragma unroll
    for (int r = 0; r < 8; ++r) { mrun[r] = -1e30f; lrun[r] = 0.0f; }

    const float scale = 0.125f;  // 1/sqrt(HD)

    issue_stage(0, 0);
    int buf = 0;
    for (int kb = 0; kb < SEQ; kb += 32) {
        wait_async<0>();       // stage(kb) landed (pre-satisfied in steady state)
        __syncthreads();       // ..for all waves; also: prior readers of buf^1 done
        if (kb + 32 < SEQ) issue_stage(kb + 32, buf ^ 1);

        // scores: 16 queries x 32 keys, reduce over hd=64 (K frags from LDS)
        v8f s0 = {}, s1 = {};
#pragma unroll
        for (int c = 0; c < 2; ++c) {
            FragB16 b0, b1;
            const unsigned short* p0 = &Ks[buf][lm][c * 32 + half * 16];
            const unsigned short* p1 = &Ks[buf][16 + lm][c * 32 + half * 16];
            b0.q[0] = *(const v4u*)p0; b0.q[1] = *(const v4u*)(p0 + 8);
            b1.q[0] = *(const v4u*)p1; b1.q[1] = *(const v4u*)(p1 + 8);
            s0 = __builtin_amdgcn_wmma_f32_16x16x32_bf16(
                false, aQ[c].v, false, b0.v, (short)0, s0, false, false);
            s1 = __builtin_amdgcn_wmma_f32_16x16x32_bf16(
                false, aQ[c].v, false, b1.v, (short)0, s1, false, false);
        }

        // streaming softmax; row r+8*half per register r
        float alpha[8], p0v[8], p1v[8];
#pragma unroll
        for (int r = 0; r < 8; ++r) {
            const float a = s0[r] * scale;
            const float b = s1[r] * scale;
            float mx = fmaxf(a, b);
#pragma unroll
            for (int off = 1; off < 16; off <<= 1)
                mx = fmaxf(mx, __shfl_xor(mx, off, 32));
            const float mn = fmaxf(mrun[r], mx);
            alpha[r] = __expf(mrun[r] - mn);
            p0v[r] = __expf(a - mn);
            p1v[r] = __expf(b - mn);
            float sm = p0v[r] + p1v[r];
#pragma unroll
            for (int off = 1; off < 16; off <<= 1)
                sm += __shfl_xor(sm, off, 32);
            lrun[r] = lrun[r] * alpha[r] + sm;
            mrun[r] = mn;
        }

        // P tile (C layout) -> wave-private LDS -> A-fragment layout.
        // Same-wave DS ops are in-order (DScnt); no block barrier required.
#pragma unroll
        for (int r = 0; r < 8; ++r) {
            pbuf[wave][r + 8 * half][lm]      = f2bf(p0v[r]);
            pbuf[wave][r + 8 * half][16 + lm] = f2bf(p1v[r]);
        }
        lds_fence();
        FragB16 aP;
        aP.q[0] = *(const v4u*)(&pbuf[wave][lm][half * 16]);
        aP.q[1] = *(const v4u*)(&pbuf[wave][lm][half * 16 + 8]);
        lds_fence();

        // O = O*alpha + P @ V   (V frags from LDS)
#pragma unroll
        for (int t = 0; t < 4; ++t) {
#pragma unroll
            for (int r = 0; r < 8; ++r) o[t][r] *= alpha[r];
            FragB16 bv;
            const unsigned short* vp = &Vs[buf][t * 16 + lm][half * 16];
            bv.q[0] = *(const v4u*)vp; bv.q[1] = *(const v4u*)(vp + 8);
            o[t] = __builtin_amdgcn_wmma_f32_16x16x32_bf16(
                false, aP.v, false, bv.v, (short)0, o[t], false, false);
        }
        buf ^= 1;
    }

    // epilogue: normalize, write [N,S,H*HD] bf16
    float invl[8];
#pragma unroll
    for (int r = 0; r < 8; ++r) invl[r] = 1.0f / lrun[r];
#pragma unroll
    for (int t = 0; t < 4; ++t) {
#pragma unroll
        for (int r = 0; r < 8; ++r) {
            const int s = qBase + r + 8 * half;
            AO[((size_t)n * SEQ + s) * EMBED + h * HD + t * 16 + lm] =
                f2bf(o[t][r] * invl[r]);
        }
    }
}

// ---------------- Output projection: out = AO @ fc_w^T + fc_b (f32) ----------------
__global__ void __launch_bounds__(256)
k_proj_gemm(const unsigned short* __restrict__ A,   // [4096,1024] bf16
            const unsigned short* __restrict__ W,   // [1024,1024] bf16
            const float* __restrict__ bias,         // [1024]
            float* __restrict__ out)                // [4096,1024] f32
{
    extern __shared__ __align__(16) unsigned short Bp[];   // [64][BROW_PAD]

    const int wave = threadIdx.x >> 5;
    const int lane = threadIdx.x & 31;
    const int half = lane >> 4;
    const int lm   = lane & 15;

    const int mBlk = blockIdx.x & 31;       // 32
    const int nBlk = blockIdx.x >> 5;       // 16
    const int m0 = mBlk * 128 + wave * 16;
    const int n0 = nBlk * 64;

    v8f acc[4] = {};
    gemm_core_128x64(A, W, Bp, m0, n0, lane, acc);

#pragma unroll
    for (int t = 0; t < 4; ++t) {
        const int e = n0 + t * 16 + lm;
        const float bv = bias[e];
#pragma unroll
        for (int r = 0; r < 8; ++r) {
            const int m = m0 + r + 8 * half;
            out[(size_t)m * EMBED + e] = acc[t][r] + bv;
        }
    }
}

// ---------------- host launcher ----------------
extern "C" void kernel_launch(void* const* d_in, const int* in_sizes, int n_in,
                              void* d_out, int out_size, void* d_ws, size_t ws_size,
                              hipStream_t stream) {
    const float* x         = (const float*)d_in[0];  // [2,2048,1024]
    const float* head_mask = (const float*)d_in[1];  // [16]
    const float* qkv_w     = (const float*)d_in[2];  // [3072,1024]
    const float* qkv_b     = (const float*)d_in[3];  // [3072]
    const float* fc_w      = (const float*)d_in[4];  // [1024,1024]
    const float* fc_b      = (const float*)d_in[5];  // [1024]
    float* out = (float*)d_out;

    // workspace layout (all bf16 ushorts): 48 MB total, < 192 MB L2
    unsigned short* Xbf = (unsigned short*)d_ws;
    unsigned short* Wbf = Xbf + (size_t)MROWS * EMBED;
    unsigned short* Fbf = Wbf + (size_t)E3 * EMBED;
    unsigned short* Qb  = Fbf + (size_t)EMBED * EMBED;
    unsigned short* Kb  = Qb  + (size_t)BATCH * HEADS * SEQ * HD;
    unsigned short* Vt  = Kb  + (size_t)BATCH * HEADS * SEQ * HD;
    unsigned short* AO  = Vt  + (size_t)BATCH * HEADS * SEQ * HD;

    const int nx4 = (MROWS * EMBED) / 4;
    const int nw4 = (E3 * EMBED) / 4;
    const int nf4 = (EMBED * EMBED) / 4;
    k_cvt_bf16<<<(nx4 + 255) / 256, 256, 0, stream>>>(x, Xbf, nx4);
    k_cvt_bf16<<<(nw4 + 255) / 256, 256, 0, stream>>>(qkv_w, Wbf, nw4);
    k_cvt_bf16<<<(nf4 + 255) / 256, 256, 0, stream>>>(fc_w, Fbf, nf4);

    const size_t panelBytes = (size_t)64 * BROW_PAD * sizeof(unsigned short); // 129KB

    // QKV GEMM: 32 M-blocks * 48 N-blocks
    k_qkv_gemm<<<32 * 48, 256, panelBytes, stream>>>(Xbf, Wbf, qkv_b, head_mask,
                                                     Qb, Kb, Vt);

    // attention: (B*H=32) * (SEQ/64=32) = 1024 blocks of 4 waves
    k_attention<<<1024, 128, 0, stream>>>(Qb, Kb, Vt, AO);

    // projection: 32 M-blocks * 16 N-blocks
    k_proj_gemm<<<32 * 16, 256, panelBytes, stream>>>(AO, Fbf, fc_b, out);
}
